// AdditiveAttention_17489106829584
// MI455X (gfx1250) — matrix operand order repaired and checked by reading, not compiled, not run on hardware
//
#include <hip/hip_runtime.h>
#include <hip/hip_bf16.h>
#include <cstdint>

// ---------------------------------------------------------------------------
// Problem constants (from the reference)
// ---------------------------------------------------------------------------
#define BB   8
#define SS   4096
#define DD   1024
#define HH   16
#define HDD  64
#define MM   (BB * SS)     // 32768 rows
#define NN3  (3 * DD)      // 3072 fused QKV columns

typedef __attribute__((ext_vector_type(16))) __bf16 v16bf;
typedef __attribute__((ext_vector_type(8)))  float  v8f;

union Frag16 {
    v16bf v;
    unsigned short u[16];
    uint4 q[2];
};

__device__ __forceinline__ unsigned short f2bf(float x) {
    unsigned int b = __float_as_uint(x);
    b += 0x7FFFu + ((b >> 16) & 1u);   // round-to-nearest-even
    return (unsigned short)(b >> 16);
}

// ---------------------------------------------------------------------------
// CDNA5 async copy: global -> LDS, tracked with ASYNCcnt (ISA 08_async_tensor).
// Builtin takes GCC-vector int4 pointers in AS1/AS3. IOFFSET is added to BOTH
// the LDS and global addresses, so one pointer pair covers 32B via {0,16}.
// ---------------------------------------------------------------------------
#if __has_builtin(__builtin_amdgcn_global_load_async_to_lds_b128) && \
    __has_builtin(__builtin_amdgcn_s_wait_asynccnt)
#define HAVE_ASYNC_LDS 1
#define GLOBAL_AS __attribute__((address_space(1)))
#define LDS_AS    __attribute__((address_space(3)))
typedef int gv4i __attribute__((vector_size(16)));
#define ASYNC_B128(gptr, lptr, off)                                          \
    __builtin_amdgcn_global_load_async_to_lds_b128(                          \
        (GLOBAL_AS gv4i*)(GLOBAL_AS void*)(gptr),                            \
        (LDS_AS gv4i*)(LDS_AS void*)(lptr), (off), 0)
#else
#define HAVE_ASYNC_LDS 0
#endif

// ---------------------------------------------------------------------------
// f32 -> bf16 conversion of X
// ---------------------------------------------------------------------------
__global__ __launch_bounds__(256) void cvt_x_kernel(const float* __restrict__ X,
                                                    unsigned short* __restrict__ Xh) {
    size_t i = ((size_t)blockIdx.x * 256 + threadIdx.x) * 4;  // MM*DD total, /4
    float4 x = *(const float4*)(X + i);
    ushort4 o;
    o.x = f2bf(x.x); o.y = f2bf(x.y); o.z = f2bf(x.z); o.w = f2bf(x.w);
    *(ushort4*)(Xh + i) = o;
}

// ---------------------------------------------------------------------------
// Tiled transpose + bf16 convert: W[k][n] (f32) -> Wt[n][k] (bf16).
// 32x32 tiles through LDS so both global reads and writes are coalesced.
// ---------------------------------------------------------------------------
__global__ __launch_bounds__(256) void cvt_w_kernel(const float* __restrict__ Wq,
                                                    const float* __restrict__ Wk,
                                                    const float* __restrict__ Wv,
                                                    unsigned short* __restrict__ Wt) {
    __shared__ unsigned short tile[32][33];
    const int mat = blockIdx.z;
    const float* W = (mat == 0) ? Wq : (mat == 1) ? Wk : Wv;
    const int n0 = blockIdx.x * 32;
    const int k0 = blockIdx.y * 32;
    const int tx = threadIdx.x & 31;
    const int ty = threadIdx.x >> 5;      // 8 rows per pass, 4 passes
#pragma unroll
    for (int p = 0; p < 4; ++p) {
        const int k = k0 + ty + 8 * p;
        tile[ty + 8 * p][tx] = f2bf(W[(size_t)k * DD + n0 + tx]);
    }
    __syncthreads();
#pragma unroll
    for (int p = 0; p < 4; ++p) {
        const int n = n0 + ty + 8 * p;
        Wt[((size_t)mat * DD + n) * DD + k0 + tx] = tile[tx][ty + 8 * p];
    }
}

__global__ __launch_bounds__(256) void cvt_wr_kernel(const float* __restrict__ Wr,
                                                     unsigned short* __restrict__ Wrt) {
    for (int i = threadIdx.x; i < HDD * HDD; i += 256) {
        int n = i >> 6, k = i & 63;
        Wrt[n * HDD + k] = f2bf(Wr[k * HDD + n]);
    }
}

// ---------------------------------------------------------------------------
// One K-step of the QKV GEMM: B fragment from global, 8 A fragments from a
// statically-addressed LDS buffer, 8 back-to-back in-place WMMAs.
// ---------------------------------------------------------------------------
__device__ __forceinline__ void gemm_step(const unsigned short (*buf)[40],
                                          const unsigned short* __restrict__ wrow,
                                          int kk, int llo, int lhi,
                                          v8f (&acc)[8]) {
    Frag16 bf;
    const uint4* wsrc = (const uint4*)(wrow + kk);
    bf.q[0] = wsrc[0];
    bf.q[1] = wsrc[2];

    Frag16 af[8];
#pragma unroll
    for (int t = 0; t < 8; ++t) {
        const int m = t * 16 + llo;
        af[t].q[0] = *(const uint4*)&buf[m][lhi * 8];
        af[t].q[1] = *(const uint4*)&buf[m][lhi * 8 + 16];
    }
#pragma unroll
    for (int t = 0; t < 8; ++t)
        acc[t] = __builtin_amdgcn_wmma_f32_16x16x32_bf16(
                     false, af[t].v, false, bf.v, (short)0, acc[t], false, false);
}

// ---------------------------------------------------------------------------
// Fused QKV GEMM:  C[32768 x 3072] = Xh * Wt^T  (+ bias), bf16 WMMA, f32 acc.
// Block = 256 thr (8 waves), tile 128x128, K-step 32.
// A tile double-buffered in LDS via async global->LDS copies.  The pipeline
// loop is manually unrolled x2 so each half addresses its buffer statically
// (keeps accumulators in place; avoids register-rotation copies).
// ---------------------------------------------------------------------------
__global__ __launch_bounds__(256) void qkv_gemm_kernel(
        const unsigned short* __restrict__ Xh,   // [MM][DD] bf16
        const unsigned short* __restrict__ Wt,   // [NN3][DD] bf16 (n-major, k contiguous)
        const float* __restrict__ bq, const float* __restrict__ bk,
        const float* __restrict__ bv,
        float* __restrict__ Q, float* __restrict__ K, float* __restrict__ V) {
    __shared__ unsigned short As[2][128][40];     // 80B row pitch: conflict-free b128 reads

    const int ntile = blockIdx.x % (NN3 / 128);
    const int mtile = blockIdx.x / (NN3 / 128);
    const int tileM = mtile * 128;
    const int tileN = ntile * 128;

    const int mat = tileN >> 10;                  // whole 128-col tile is in one matrix
    float*       Cmat = (mat == 0) ? Q  : (mat == 1) ? K  : V;
    const float* bias = (mat == 0) ? bq : (mat == 1) ? bk : bv;

    const int tid  = threadIdx.x;
    const int wave = tid >> 5;
    const int lane = tid & 31;
    const int lhi  = lane >> 4;      // 0/1: K-half selector
    const int llo  = lane & 15;

    v8f acc[8] = {};

    const int nbase = tileN + wave * 16;
    const unsigned short* wrow = Wt + (size_t)(nbase + llo) * DD + lhi * 8;

    const int arow  = tid >> 1;
    const int ahalf = (tid & 1) * 16;
    const unsigned short* xrow = Xh + (size_t)(tileM + arow) * DD + ahalf;

#if HAVE_ASYNC_LDS
    // ---- software-pipelined async staging, manual x2 unroll ----
    ASYNC_B128(xrow, &As[0][arow][ahalf], 0);
    ASYNC_B128(xrow, &As[0][arow][ahalf], 16);

    for (int step = 0; step < DD / 32; step += 2) {
        // ---- even half: compute from buffer 0, prefetch tile step+1 -> buf 1
        {
            const unsigned short* g = xrow + (step + 1) * 32;
            ASYNC_B128(g, &As[1][arow][ahalf], 0);
            ASYNC_B128(g, &As[1][arow][ahalf], 16);
            __builtin_amdgcn_s_wait_asynccnt(2);   // in-order: tile `step` done
            __syncthreads();
            gemm_step(As[0], wrow, step * 32, llo, lhi, acc);
            __syncthreads();
        }
        // ---- odd half: compute from buffer 1, prefetch tile step+2 -> buf 0
        {
            if (step + 2 < DD / 32) {
                const unsigned short* g = xrow + (step + 2) * 32;
                ASYNC_B128(g, &As[0][arow][ahalf], 0);
                ASYNC_B128(g, &As[0][arow][ahalf], 16);
                __builtin_amdgcn_s_wait_asynccnt(2);   // tile `step+1` done
            } else {
                __builtin_amdgcn_s_wait_asynccnt(0);
            }
            __syncthreads();
            gemm_step(As[1], wrow, (step + 1) * 32, llo, lhi, acc);
            __syncthreads();
        }
    }
#else
    // ---- fallback: synchronous staging through VGPRs ----
    for (int kk = 0; kk < DD; kk += 32) {
        const uint4* src = (const uint4*)(xrow + kk);
        uint4 a0 = src[0];
        uint4 a1 = src[1];
        __syncthreads();
        *(uint4*)&As[0][arow][ahalf]     = a0;
        *(uint4*)&As[0][arow][ahalf + 8] = a1;
        __syncthreads();
        gemm_step(As[0], wrow, kk, llo, lhi, acc);
    }
#endif

    // Epilogue: bias + scatter to [B,H,S,HD]
    const int d0 = (tileN & 1023) + wave * 16 + llo;
    const float bval = bias[d0];
    const int h  = d0 >> 6;
    const int hd = d0 & 63;
#pragma unroll
    for (int t = 0; t < 8; ++t) {
        const int mbase = tileM + t * 16 + lhi * 8;
#pragma unroll
        for (int vg = 0; vg < 8; ++vg) {
            const int m = mbase + vg;
            const int b = m >> 12;
            const int s = m & 4095;
            Cmat[(((size_t)(b * HH + h) * SS + s) << 6) + hd] = acc[t][vg] + bval;
        }
    }
}

// ---------------------------------------------------------------------------
// Query pooling: per (b,h) block — scalar logits, softmax over S, weighted sum
// gq[d] = sum_s alpha[s] * q[s][d]
// ---------------------------------------------------------------------------
__global__ __launch_bounds__(256) void pool_q_kernel(const float* __restrict__ Q,
                                                     const float* __restrict__ wql,
                                                     const float* __restrict__ bql,
                                                     float* __restrict__ gq) {
    __shared__ float w[SS];
    __shared__ float wv[HDD];
    __shared__ float red[256];
    const int bh  = blockIdx.x;
    const int tid = threadIdx.x;
    const float* Qb = Q + (size_t)bh * SS * HDD;

    if (tid < HDD) wv[tid] = wql[tid];
    __syncthreads();

    const float scale = 0.125f;  // 1/sqrt(64)
    float lmax = -3.4e38f;
    for (int s = tid; s < SS; s += 256) {
        const float* row = Qb + s * HDD;
        float d = 0.f;
#pragma unroll 8
        for (int j = 0; j < HDD; ++j) d += row[j] * wv[j];
        d = (d + bql[0]) * scale;
        w[s] = d;
        lmax = fmaxf(lmax, d);
    }
    red[tid] = lmax;
    __syncthreads();
    for (int off = 128; off > 0; off >>= 1) {
        if (tid < off) red[tid] = fmaxf(red[tid], red[tid + off]);
        __syncthreads();
    }
    const float mx = red[0];
    __syncthreads();

    float lsum = 0.f;
    for (int s = tid; s < SS; s += 256) {
        float e = __expf(w[s] - mx);
        w[s] = e;
        lsum += e;
    }
    red[tid] = lsum;
    __syncthreads();
    for (int off = 128; off > 0; off >>= 1) {
        if (tid < off) red[tid] += red[tid + off];
        __syncthreads();
    }
    const float inv = 1.f / red[0];
    __syncthreads();

    const int d = tid & 63, g = tid >> 6;   // 4 groups of 64
    float acc = 0.f;
    for (int s = g; s < SS; s += 4) acc += w[s] * Qb[s * HDD + d];
    red[tid] = acc;
    __syncthreads();
    if (tid < HDD) {
        float r = red[tid] + red[tid + 64] + red[tid + 128] + red[tid + 192];
        gq[bh * HDD + tid] = r * inv;
    }
}

// ---------------------------------------------------------------------------
// Key pooling with gating:  p = k * gq;  beta = softmax((p.wkl+bkl)/8)
// gk[d] = gq[d] * sum_s beta[s] * k[s][d]
// ---------------------------------------------------------------------------
__global__ __launch_bounds__(256) void pool_k_kernel(const float* __restrict__ Kt,
                                                     const float* __restrict__ gq,
                                                     const float* __restrict__ wkl,
                                                     const float* __restrict__ bkl,
                                                     float* __restrict__ gk) {
    __shared__ float w[SS];
    __shared__ float wg[HDD];
    __shared__ float gql[HDD];
    __shared__ float red[256];
    const int bh  = blockIdx.x;
    const int tid = threadIdx.x;
    const float* Kb = Kt + (size_t)bh * SS * HDD;

    if (tid < HDD) {
        float g = gq[bh * HDD + tid];
        gql[tid] = g;
        wg[tid]  = g * wkl[tid];
    }
    __syncthreads();

    const float scale = 0.125f;
    float lmax = -3.4e38f;
    for (int s = tid; s < SS; s += 256) {
        const float* row = Kb + s * HDD;
        float d = 0.f;
#pragma unroll 8
        for (int j = 0; j < HDD; ++j) d += row[j] * wg[j];
        d = (d + bkl[0]) * scale;
        w[s] = d;
        lmax = fmaxf(lmax, d);
    }
    red[tid] = lmax;
    __syncthreads();
    for (int off = 128; off > 0; off >>= 1) {
        if (tid < off) red[tid] = fmaxf(red[tid], red[tid + off]);
        __syncthreads();
    }
    const float mx = red[0];
    __syncthreads();

    float lsum = 0.f;
    for (int s = tid; s < SS; s += 256) {
        float e = __expf(w[s] - mx);
        w[s] = e;
        lsum += e;
    }
    red[tid] = lsum;
    __syncthreads();
    for (int off = 128; off > 0; off >>= 1) {
        if (tid < off) red[tid] += red[tid + off];
        __syncthreads();
    }
    const float inv = 1.f / red[0];
    __syncthreads();

    const int d = tid & 63, g = tid >> 6;
    float acc = 0.f;
    for (int s = g; s < SS; s += 4) acc += w[s] * Kb[s * HDD + d];
    red[tid] = acc;
    __syncthreads();
    if (tid < HDD) {
        float r = red[tid] + red[tid + 64] + red[tid + 128] + red[tid + 192];
        gk[bh * HDD + tid] = r * inv * gql[tid];
    }
}

// ---------------------------------------------------------------------------
// Output: out[b,s,h*64+n] = q + (v .* gk) @ Wr + br
// Block = 256 thr (8 waves); each wave: 16 rows x 64 cols via 4x(2 K-step) WMMA.
// u fragments built on the fly (f32 load, gate, bf16 pack).
// ---------------------------------------------------------------------------
__global__ __launch_bounds__(256) void out_proj_kernel(
        const float* __restrict__ Q, const float* __restrict__ V,
        const float* __restrict__ gk, const unsigned short* __restrict__ Wrt,
        const float* __restrict__ br, float* __restrict__ out) {
    __shared__ float gks[HDD];
    const int blk = blockIdx.x;           // ((b*16+h)*32 + row_tile)
    const int rt  = blk & 31;
    const int bh  = blk >> 5;
    const int b   = bh >> 4;
    const int h   = bh & 15;
    const int tid  = threadIdx.x;
    const int wave = tid >> 5;
    const int lane = tid & 31;
    const int lhi  = lane >> 4;
    const int llo  = lane & 15;

    if (tid < HDD) gks[tid] = gk[bh * HDD + tid];
    __syncthreads();

    const int r0 = rt * 128 + wave * 16;
    const float* Vb = V + ((size_t)bh * SS + r0) * HDD;
    const float* Qb = Q + ((size_t)bh * SS + r0) * HDD;

    v8f acc[4] = {};

#pragma unroll
    for (int ks = 0; ks < 2; ++ks) {
        const int kb = ks * 32 + lhi * 8;
        const float* vrow = Vb + llo * HDD;
        float4 x0 = *(const float4*)(vrow + kb);
        float4 x1 = *(const float4*)(vrow + kb + 4);
        float4 x2 = *(const float4*)(vrow + kb + 16);
        float4 x3 = *(const float4*)(vrow + kb + 20);

        float va[16];
        va[0] = x0.x; va[1] = x0.y; va[2]  = x0.z; va[3]  = x0.w;
        va[4] = x1.x; va[5] = x1.y; va[6]  = x1.z; va[7]  = x1.w;
        va[8] = x2.x; va[9] = x2.y; va[10] = x2.z; va[11] = x2.w;
        va[12] = x3.x; va[13] = x3.y; va[14] = x3.z; va[15] = x3.w;

        Frag16 af;
#pragma unroll
        for (int j = 0; j < 8; ++j) af.u[j]     = f2bf(va[j]     * gks[kb + j]);
#pragma unroll
        for (int j = 0; j < 8; ++j) af.u[8 + j] = f2bf(va[8 + j] * gks[kb + 16 + j]);

#pragma unroll
        for (int nt = 0; nt < 4; ++nt) {
            Frag16 bf;
            const unsigned short* wr = Wrt + (nt * 16 + llo) * HDD + ks * 32 + lhi * 8;
            bf.q[0] = *(const uint4*)wr;
            bf.q[1] = *(const uint4*)(wr + 16);
            acc[nt] = __builtin_amdgcn_wmma_f32_16x16x32_bf16(
                          false, af.v, false, bf.v, (short)0, acc[nt], false, false);
        }
    }

#pragma unroll
    for (int nt = 0; nt < 4; ++nt) {
        const int col = nt * 16 + llo;
        const float bb = br[col];
#pragma unroll
        for (int vg = 0; vg < 8; ++vg) {
            const int rloc = vg + lhi * 8;
            const int s = r0 + rloc;
            out[((size_t)(b * SS + s)) * DD + h * HDD + col] =
                acc[nt][vg] + Qb[rloc * HDD + col] + bb;
        }
    }
}

// ---------------------------------------------------------------------------
// Launch
// ---------------------------------------------------------------------------
extern "C" void kernel_launch(void* const* d_in, const int* in_sizes, int n_in,
                              void* d_out, int out_size, void* d_ws, size_t ws_size,
                              hipStream_t stream) {
    const float* X   = (const float*)d_in[0];
    const float* Wq  = (const float*)d_in[1];
    const float* bq  = (const float*)d_in[2];
    const float* Wk  = (const float*)d_in[3];
    const float* bk  = (const float*)d_in[4];
    const float* Wv  = (const float*)d_in[5];
    const float* bv  = (const float*)d_in[6];
    const float* wql = (const float*)d_in[7];
    const float* bql = (const float*)d_in[8];
    const float* wkl = (const float*)d_in[9];
    const float* bkl = (const float*)d_in[10];
    const float* Wr  = (const float*)d_in[11];
    const float* br  = (const float*)d_in[12];
    float* out = (float*)d_out;

    // Workspace layout (bytes)
    char* ws = (char*)d_ws;
    size_t off = 0;
    unsigned short* Xh  = (unsigned short*)(ws + off); off += (size_t)MM * DD * 2;     // 64 MB
    unsigned short* Wt  = (unsigned short*)(ws + off); off += (size_t)NN3 * DD * 2;    // 6 MB
    unsigned short* Wrt = (unsigned short*)(ws + off); off += (size_t)HDD * HDD * 2;   // 8 KB
    off = (off + 255) & ~(size_t)255;
    float* Qbuf = (float*)(ws + off); off += (size_t)MM * DD * 4;                      // 128 MB
    float* Kbuf = (float*)(ws + off); off += (size_t)MM * DD * 4;                      // 128 MB
    float* Vbuf = (float*)(ws + off); off += (size_t)MM * DD * 4;                      // 128 MB
    float* gq   = (float*)(ws + off); off += (size_t)BB * HH * HDD * 4;
    float* gk   = (float*)(ws + off); off += (size_t)BB * HH * HDD * 4;

    // 1) precision conversion / weight transposes
    cvt_x_kernel<<<(MM * (size_t)DD) / (256 * 4), 256, 0, stream>>>(X, Xh);
    cvt_w_kernel<<<dim3(DD / 32, DD / 32, 3), 256, 0, stream>>>(Wq, Wk, Wv, Wt);
    cvt_wr_kernel<<<1, 256, 0, stream>>>(Wr, Wrt);

    // 2) fused QKV GEMM (bf16 WMMA, async LDS pipeline) -> head-major Q/K/V
    qkv_gemm_kernel<<<(MM / 128) * (NN3 / 128), 256, 0, stream>>>(
        Xh, Wt, bq, bk, bv, Qbuf, Kbuf, Vbuf);

    // 3) additive-attention poolings
    pool_q_kernel<<<BB * HH, 256, 0, stream>>>(Qbuf, wql, bql, gq);
    pool_k_kernel<<<BB * HH, 256, 0, stream>>>(Kbuf, gq, wkl, bkl, gk);

    // 4) gated value, 64x64 projection (bf16 WMMA), bias + residual, un-split heads
    out_proj_kernel<<<BB * HH * (SS / 128), 256, 0, stream>>>(
        Qbuf, Vbuf, gk, Wrt, br, out);
}
